// ViTT_2559800509062
// MI455X (gfx1250) — compile-verified, matrix operand order checked by reading, not run on hardware
//
#include <hip/hip_runtime.h>
#include <hip/hip_bf16.h>

// Recurrent transformer-decoder on MI455X (gfx1250, wave32).
// Loop-invariant self-attention + cross-attn Q hoisted out of the 128-step
// recurrence. All GEMMs run on V_WMMA_F32_16X16X32_BF16 (f32 accumulate),
// register-blocked 32x32 per wave (4 independent accumulators, 4 WMMA/k-step).
// Weights converted once to bf16 and transposed to [N][K] so A and B WMMA
// fragments are both two contiguous 16B loads per lane.

#define BROWS 128
#define DMODEL 512

typedef __attribute__((ext_vector_type(16))) __bf16 bf16x16;
typedef __attribute__((ext_vector_type(8)))  __bf16 bf16x8;
typedef __attribute__((ext_vector_type(8)))  float  f32x8;
typedef int vsi4 __attribute__((vector_size(16)));   // b128 unit for async-LDS builtin

static __device__ __forceinline__ bf16x16 load_frag(const __bf16* __restrict__ p) {
  // lane r (0-15): row r, K = [0..7] and [16..23]; lane 16+r: row r, K = [8..15],[24..31]
  bf16x8 lo = *(const bf16x8*)p;
  bf16x8 hi = *(const bf16x8*)(p + 16);
  bf16x16 r;
#pragma unroll
  for (int i = 0; i < 8; ++i) { r[i] = lo[i]; r[i + 8] = hi[i]; }
  return r;
}

// Generic batched GEMM: C[b] = A[b](MxK) * Bt[b](NxK)^T, epilogue fusions.
// One wave computes a 32x32 output macro-tile (2x2 WMMA tiles).
__global__ __launch_bounds__(256) void gemm_bf16_wmma(
    const __bf16* __restrict__ A, int lda, int strideA,
    const __bf16* __restrict__ Bt, int ldb, int strideB,
    const float* __restrict__ bias,
    const float* __restrict__ resid, int ldr,
    float* __restrict__ Cf, int ldcf, int strideCf,
    __bf16* __restrict__ Cb, int ldcb, int strideCb,
    __bf16* __restrict__ Ct, int ldct,
    int M, int N, int K, float scale, int relu)
{
  const int b    = blockIdx.z;
  const int wave = threadIdx.x >> 5;
  const int lane = threadIdx.x & 31;
  const int nt2  = N >> 5;
  const int tile = blockIdx.x * 8 + wave;
  if (tile >= (M >> 5) * nt2) return;          // wave-uniform: EXEC stays all-1s
  const int mt = (tile / nt2) << 5;
  const int nt = (tile % nt2) << 5;

  const int rsel = lane & 15;
  const int koff = (lane & 16) ? 8 : 0;        // also the M offset of lanes 16-31 in C/D
  const __bf16* pA0 = A + (size_t)b * strideA + (size_t)(mt + rsel) * lda + koff;
  const __bf16* pA1 = pA0 + (size_t)16 * lda;
  const __bf16* pB0 = Bt + (size_t)b * strideB + (size_t)(nt + rsel) * ldb + koff;
  const __bf16* pB1 = pB0 + (size_t)16 * ldb;

  f32x8 c00 = {}, c01 = {}, c10 = {}, c11 = {};
  for (int k0 = 0; k0 < K; k0 += 32) {
    bf16x16 a0 = load_frag(pA0 + k0);
    bf16x16 a1 = load_frag(pA1 + k0);
    bf16x16 b0 = load_frag(pB0 + k0);
    bf16x16 b1 = load_frag(pB1 + k0);
    c00 = __builtin_amdgcn_wmma_f32_16x16x32_bf16(false, a0, false, b0, (short)0, c00, false, false);
    c01 = __builtin_amdgcn_wmma_f32_16x16x32_bf16(false, a0, false, b1, (short)0, c01, false, false);
    c10 = __builtin_amdgcn_wmma_f32_16x16x32_bf16(false, a1, false, b0, (short)0, c10, false, false);
    c11 = __builtin_amdgcn_wmma_f32_16x16x32_bf16(false, a1, false, b1, (short)0, c11, false, false);
  }

  f32x8 acc[2][2] = {{c00, c01}, {c10, c11}};
#pragma unroll
  for (int i = 0; i < 2; ++i) {
#pragma unroll
    for (int j = 0; j < 2; ++j) {
      const int n = nt + j * 16 + rsel;
      const float bv = bias ? bias[n] : 0.f;
#pragma unroll
      for (int v = 0; v < 8; ++v) {
        const int m = mt + i * 16 + koff + v;
        float val = acc[i][j][v] * scale + bv;
        if (resid) val += resid[(size_t)m * ldr + n];
        if (relu)  val = fmaxf(val, 0.f);
        if (Cf) Cf[(size_t)b * strideCf + (size_t)m * ldcf + n] = val;
        if (Cb) Cb[(size_t)b * strideCb + (size_t)m * ldcb + n] = (__bf16)val;
        if (Ct) Ct[(size_t)n * ldct + m] = (__bf16)val;   // transposed store (for V^T)
      }
    }
  }
}

// softmax over rows of length 128 (8 heads * 128 q-rows = 1024 blocks)
__global__ __launch_bounds__(128) void softmax_kernel(
    const float* __restrict__ S, __bf16* __restrict__ P, float scale)
{
  __shared__ float red[128];
  const int row = blockIdx.x, tid = threadIdx.x;
  float v = S[(size_t)row * 128 + tid] * scale;
  red[tid] = v; __syncthreads();
  for (int s = 64; s > 0; s >>= 1) { if (tid < s) red[tid] = fmaxf(red[tid], red[tid + s]); __syncthreads(); }
  const float mx = red[0]; __syncthreads();
  const float e = __expf(v - mx);
  red[tid] = e; __syncthreads();
  for (int s = 64; s > 0; s >>= 1) { if (tid < s) red[tid] += red[tid + s]; __syncthreads(); }
  P[(size_t)row * 128 + tid] = (__bf16)(e * (1.f / red[0]));
}

// LayerNorm over rows of 512; optional f32/bf16 outputs, optional r += y update
// (writes r, bf16(r), and outs row for row 0).
__global__ __launch_bounds__(256) void layernorm_kernel(
    const float* __restrict__ t, const float* __restrict__ g, const float* __restrict__ be,
    float* __restrict__ outF, __bf16* __restrict__ outB,
    float* __restrict__ r, __bf16* __restrict__ rB, __bf16* __restrict__ outsRow)
{
  __shared__ float s1[256], s2[256];
  const int row = blockIdx.x, tid = threadIdx.x;
  const float* tr = t + (size_t)row * DMODEL;
  const float a0 = tr[tid], a1 = tr[tid + 256];
  s1[tid] = a0 + a1; s2[tid] = a0 * a0 + a1 * a1;
  __syncthreads();
  for (int s = 128; s > 0; s >>= 1) {
    if (tid < s) { s1[tid] += s1[tid + s]; s2[tid] += s2[tid + s]; }
    __syncthreads();
  }
  const float mean = s1[0] * (1.f / DMODEL);
  const float var  = s2[0] * (1.f / DMODEL) - mean * mean;
  const float rstd = rsqrtf(var + 1e-5f);
  for (int i = tid; i < DMODEL; i += 256) {
    const float y = (tr[i] - mean) * rstd * g[i] + be[i];
    if (outF) outF[(size_t)row * DMODEL + i] = y;
    if (outB) outB[(size_t)row * DMODEL + i] = (__bf16)y;
    if (r) {
      const float nr = r[(size_t)row * DMODEL + i] + y;
      r[(size_t)row * DMODEL + i]  = nr;
      rB[(size_t)row * DMODEL + i] = (__bf16)nr;
      if (row == 0 && outsRow) outsRow[i] = (__bf16)nr;
    }
  }
}

// f32 W[K][N] -> bf16 Wt[N][K] (once per launch; weights then live in L2).
// Streaming read with explicit prefetch (global_prefetch_b8 on gfx1250).
__global__ __launch_bounds__(256) void trans_cvt(const float* __restrict__ W,
                                                 __bf16* __restrict__ Wt, int K, int N)
{
  const int idx = blockIdx.x * 256 + threadIdx.x;
  if (idx >= K * N) return;
  if (idx + 4096 < K * N) __builtin_prefetch(W + idx + 4096, 0, 1);
  const int k = idx / N, n = idx - k * N;
  Wt[(size_t)n * K + k] = (__bf16)W[idx];
}

// x (f32) -> bf16, staged through LDS with gfx1250 async global->LDS copy when
// the toolchain exposes the builtins (falls back to direct loads otherwise).
__global__ __launch_bounds__(256) void cvt_bf16_async(const float* __restrict__ in,
                                                      __bf16* __restrict__ out)
{
  __shared__ float buf[1024];
  const int tid  = threadIdx.x;
  const int base = blockIdx.x * 1024;
#if defined(__AMDGCN__) && \
    __has_builtin(__builtin_amdgcn_global_load_async_to_lds_b128) && \
    __has_builtin(__builtin_amdgcn_s_wait_asynccnt)
  {
    // each thread async-copies one b128 (4 floats) global -> LDS
    __builtin_amdgcn_global_load_async_to_lds_b128(
        (__attribute__((address_space(1))) vsi4*)(in + base + tid * 4),
        (__attribute__((address_space(3))) vsi4*)(buf + tid * 4),
        0, 0);
    __builtin_amdgcn_s_wait_asynccnt(0);
    __syncthreads();
  }
#else
  buf[tid]       = in[base + tid];
  buf[tid + 256] = in[base + tid + 256];
  buf[tid + 512] = in[base + tid + 512];
  buf[tid + 768] = in[base + tid + 768];
  __syncthreads();
#endif
#pragma unroll
  for (int j = 0; j < 4; ++j) {
    const int i = tid + j * 256;
    out[base + i] = (__bf16)buf[i];
  }
}

__global__ __launch_bounds__(256) void zero_rr(float* __restrict__ r,
                                               __bf16* __restrict__ rb, int n)
{
  const int i = blockIdx.x * 256 + threadIdx.x;
  if (i < n) { r[i] = 0.f; rb[i] = (__bf16)0.f; }
}

extern "C" void kernel_launch(void* const* d_in, const int* in_sizes, int n_in,
                              void* d_out, int out_size, void* d_ws, size_t ws_size,
                              hipStream_t stream)
{
  (void)in_sizes; (void)n_in; (void)out_size; (void)ws_size;
  const float* x     = (const float*)d_in[0];
  const float* saWq  = (const float*)d_in[1];
  const float* saWk  = (const float*)d_in[2];
  const float* saWv  = (const float*)d_in[3];
  const float* sabq  = (const float*)d_in[4];
  const float* sabk  = (const float*)d_in[5];
  const float* sabv  = (const float*)d_in[6];
  const float* saWo  = (const float*)d_in[7];
  const float* sabo  = (const float*)d_in[8];
  const float* caWq  = (const float*)d_in[9];
  const float* caWk  = (const float*)d_in[10];
  const float* caWv  = (const float*)d_in[11];
  const float* cabq  = (const float*)d_in[12];
  const float* cabk  = (const float*)d_in[13];
  const float* cabv  = (const float*)d_in[14];
  const float* caWo  = (const float*)d_in[15];
  const float* cabo  = (const float*)d_in[16];
  const float* W1    = (const float*)d_in[17];
  const float* b1    = (const float*)d_in[18];
  const float* W2    = (const float*)d_in[19];
  const float* b2    = (const float*)d_in[20];
  const float* g1    = (const float*)d_in[21];
  const float* be1   = (const float*)d_in[22];
  const float* g2    = (const float*)d_in[23];
  const float* be2   = (const float*)d_in[24];
  const float* g3    = (const float*)d_in[25];
  const float* be3   = (const float*)d_in[26];
  const float* Wr1   = (const float*)d_in[27];
  const float* br1   = (const float*)d_in[28];
  const float* Wr2   = (const float*)d_in[29];
  const float* br2   = (const float*)d_in[30];

  size_t off = 0;
  auto allocB = [&](size_t elems) -> __bf16* {
    off = (off + 255) & ~(size_t)255;
    __bf16* p = (__bf16*)((char*)d_ws + off); off += elems * sizeof(__bf16); return p;
  };
  auto allocF = [&](size_t elems) -> float* {
    off = (off + 255) & ~(size_t)255;
    float* p = (float*)((char*)d_ws + off); off += elems * sizeof(float); return p;
  };

  // bf16 transposed weights [N][K]
  __bf16* WqTsa = allocB(512 * 512);
  __bf16* WkTsa = allocB(512 * 512);
  __bf16* WvTsa = allocB(512 * 512);
  __bf16* WoTsa = allocB(512 * 512);
  __bf16* WqTca = allocB(512 * 512);
  __bf16* WkTca = allocB(512 * 512);
  __bf16* WvTca = allocB(512 * 512);
  __bf16* WoTca = allocB(512 * 512);
  __bf16* W1T   = allocB(256 * 512);
  __bf16* W2T   = allocB(512 * 256);
  __bf16* Wr1T  = allocB(512 * 512);
  __bf16* Wr2T  = allocB(1024 * 512);
  // activations
  __bf16* x_b   = allocB(BROWS * DMODEL);
  float*  x1    = allocF(BROWS * DMODEL);
  __bf16* x1_b  = allocB(BROWS * DMODEL);
  __bf16* qca   = allocB(BROWS * DMODEL);
  float*  r     = allocF(BROWS * DMODEL);
  __bf16* r_b   = allocB(BROWS * DMODEL);
  __bf16* Qs    = allocB(BROWS * DMODEL);
  __bf16* Kp    = allocB(BROWS * DMODEL);
  __bf16* Vt    = allocB(DMODEL * BROWS);       // transposed V
  float*  S     = allocF(8 * 128 * 128);
  __bf16* P     = allocB(8 * 128 * 128);
  __bf16* O     = allocB(BROWS * DMODEL);
  float*  t     = allocF(BROWS * DMODEL);
  float*  x2    = allocF(BROWS * DMODEL);
  __bf16* x2_b  = allocB(BROWS * DMODEL);
  __bf16* ffh   = allocB(BROWS * 256);
  __bf16* outsb = allocB(BROWS * DMODEL);
  __bf16* h1_b  = allocB(BROWS * DMODEL);

  auto trans = [&](const float* W, __bf16* Wt, int K, int N) {
    int n = K * N;
    trans_cvt<<<(n + 255) / 256, 256, 0, stream>>>(W, Wt, K, N);
  };
  trans(saWq, WqTsa, 512, 512);  trans(saWk, WkTsa, 512, 512);
  trans(saWv, WvTsa, 512, 512);  trans(saWo, WoTsa, 512, 512);
  trans(caWq, WqTca, 512, 512);  trans(caWk, WkTca, 512, 512);
  trans(caWv, WvTca, 512, 512);  trans(caWo, WoTca, 512, 512);
  trans(W1,   W1T,   512, 256);  trans(W2,   W2T,   256, 512);
  trans(Wr1,  Wr1T,  512, 512);  trans(Wr2,  Wr2T,  512, 1024);

  cvt_bf16_async<<<(BROWS * DMODEL) / 1024, 256, 0, stream>>>(x, x_b);
  zero_rr<<<(BROWS * DMODEL + 255) / 256, 256, 0, stream>>>(r, r_b, BROWS * DMODEL);

  auto gemm = [&](const __bf16* A, int lda, int sA,
                  const __bf16* Bt, int ldb, int sB,
                  const float* bias, const float* resid, int ldr,
                  float* Cf, int ldcf, int sCf,
                  __bf16* Cb, int ldcb, int sCb,
                  __bf16* Ct, int ldct,
                  int M, int N, int K, float scale, int relu, int batch) {
    int tiles = (M / 32) * (N / 32);
    dim3 grid((tiles + 7) / 8, 1, batch);
    gemm_bf16_wmma<<<grid, 256, 0, stream>>>(A, lda, sA, Bt, ldb, sB, bias, resid, ldr,
                                             Cf, ldcf, sCf, Cb, ldcb, sCb, Ct, ldct,
                                             M, N, K, scale, relu);
  };

  // ---------- loop-invariant: self-attention, x1, cross-attn Q ----------
  gemm(x_b, 512, 0, WqTsa, 512, 0, sabq, nullptr, 0, nullptr, 0, 0, Qs, 512, 0, nullptr, 0, 128, 512, 512, 1.f, 0, 1);
  gemm(x_b, 512, 0, WkTsa, 512, 0, sabk, nullptr, 0, nullptr, 0, 0, Kp, 512, 0, nullptr, 0, 128, 512, 512, 1.f, 0, 1);
  gemm(x_b, 512, 0, WvTsa, 512, 0, sabv, nullptr, 0, nullptr, 0, 0, nullptr, 0, 0, Vt, 128, 128, 512, 512, 1.f, 0, 1);
  gemm(Qs, 512, 64, Kp, 512, 64, nullptr, nullptr, 0, S, 128, 16384, nullptr, 0, 0, nullptr, 0, 128, 128, 64, 1.f, 0, 8);
  softmax_kernel<<<1024, 128, 0, stream>>>(S, P, 0.125f);
  gemm(P, 128, 16384, Vt, 128, 8192, nullptr, nullptr, 0, nullptr, 0, 0, O, 512, 64, nullptr, 0, 128, 64, 128, 1.f, 0, 8);
  gemm(O, 512, 0, WoTsa, 512, 0, sabo, x, 512, t, 512, 0, nullptr, 0, 0, nullptr, 0, 128, 512, 512, 1.f, 0, 1);
  layernorm_kernel<<<128, 256, 0, stream>>>(t, g1, be1, x1, x1_b, nullptr, nullptr, nullptr);
  gemm(x1_b, 512, 0, WqTca, 512, 0, cabq, nullptr, 0, nullptr, 0, 0, qca, 512, 0, nullptr, 0, 128, 512, 512, 1.f, 0, 1);

  // ---------- 128-step recurrence ----------
  for (int s = 0; s < BROWS; ++s) {
    gemm(r_b, 512, 0, WkTca, 512, 0, cabk, nullptr, 0, nullptr, 0, 0, Kp, 512, 0, nullptr, 0, 128, 512, 512, 1.f, 0, 1);
    gemm(r_b, 512, 0, WvTca, 512, 0, cabv, nullptr, 0, nullptr, 0, 0, nullptr, 0, 0, Vt, 128, 128, 512, 512, 1.f, 0, 1);
    gemm(qca, 512, 64, Kp, 512, 64, nullptr, nullptr, 0, S, 128, 16384, nullptr, 0, 0, nullptr, 0, 128, 128, 64, 1.f, 0, 8);
    softmax_kernel<<<1024, 128, 0, stream>>>(S, P, 0.125f);
    gemm(P, 128, 16384, Vt, 128, 8192, nullptr, nullptr, 0, nullptr, 0, 0, O, 512, 64, nullptr, 0, 128, 64, 128, 1.f, 0, 8);
    gemm(O, 512, 0, WoTca, 512, 0, cabo, x1, 512, t, 512, 0, nullptr, 0, 0, nullptr, 0, 128, 512, 512, 1.f, 0, 1);
    layernorm_kernel<<<128, 256, 0, stream>>>(t, g2, be2, x2, x2_b, nullptr, nullptr, nullptr);
    gemm(x2_b, 512, 0, W1T, 512, 0, b1, nullptr, 0, nullptr, 0, 0, ffh, 256, 0, nullptr, 0, 128, 256, 512, 1.f, 1, 1);
    gemm(ffh, 256, 0, W2T, 256, 0, b2, x2, 512, t, 512, 0, nullptr, 0, 0, nullptr, 0, 128, 512, 256, 1.f, 0, 1);
    layernorm_kernel<<<128, 256, 0, stream>>>(t, g3, be3, nullptr, nullptr, r, r_b, outsb + (size_t)s * DMODEL);
  }

  // ---------- output head ----------
  gemm(outsb, 512, 0, Wr1T, 512, 0, br1, nullptr, 0, nullptr, 0, 0, h1_b, 512, 0, nullptr, 0, 128, 512, 512, 1.f, 0, 1);
  gemm(h1_b, 512, 0, Wr2T, 512, 0, br2, nullptr, 0, (float*)d_out, 1024, 0, nullptr, 0, 0, nullptr, 0, 128, 1024, 512, 1.f, 0, 1);
}
